// Separator_9655086481591
// MI455X (gfx1250) — compile-verified, matrix operand order checked by reading, not run on hardware
//
#include <hip/hip_runtime.h>
#include <hip/hip_bf16.h>
#include <math.h>

#define N_FFT    4096
#define LOG2N    12
#define HOP      1024
#define F_BINS   2049
#define N_SRC    4
#define N_CH     2
#define WIN_LEN  300
#define EPS_F    1e-10f
#define SQRT_EPS 1e-5f
#define SCALE_F  10.0f
#define PI_F     3.14159265358979f

typedef __attribute__((ext_vector_type(16))) __bf16 v16bf;
typedef __attribute__((ext_vector_type(8)))  __bf16 v8bf;
typedef __attribute__((ext_vector_type(8)))  float  v8f;

__device__ __forceinline__ float hann_w(int n) {
  return 0.5f * (1.0f - __cosf((2.0f * PI_F / (float)N_FFT) * (float)n));
}
__device__ __forceinline__ int brev12(int x) {
  return (int)(__brev((unsigned)x) >> (32 - LOG2N));
}
__device__ __forceinline__ float2 cmul(float2 a, float2 b) {
  return make_float2(a.x * b.x - a.y * b.y, a.x * b.y + a.y * b.x);
}

// -------- shared-memory radix-2 DIT FFT, twiddles from LDS table ----------
// tw[k] = exp(-i*pi*k/2048); sign=+1 -> forward, sign=-1 -> inverse (unscaled)
__device__ void fft_lds(float2* sh, const float2* tw, int tid, int nthr, float sign) {
  for (int stage = 0; stage < LOG2N; ++stage) {
    int half  = 1 << stage;
    int shift = (LOG2N - 1) - stage;
    for (int b = tid; b < (N_FFT / 2); b += nthr) {
      int j  = b & (half - 1);
      int i0 = ((b >> stage) << (stage + 1)) + j;
      int i1 = i0 + half;
      float2 w  = tw[j << shift];
      float2 wz = make_float2(w.x, sign * w.y);
      float2 u  = sh[i0];
      float2 t  = cmul(sh[i1], wz);
      sh[i0] = make_float2(u.x + t.x, u.y + t.y);
      sh[i1] = make_float2(u.x - t.x, u.y - t.y);
    }
    __syncthreads();
  }
}

__device__ void init_twiddles(float2* tw, int tid, int nthr) {
  for (int k = tid; k < N_FFT / 2; k += nthr) {
    float sn, cs;
    __sincosf(PI_F * (float)k / (float)(N_FFT / 2), &sn, &cs);
    tw[k] = make_float2(cs, -sn);
  }
}

// -------------------- 1) STFT: audio -> mix (complex), mag ----------------
// mix/mag layout: [c][f][t]
__global__ void stft_kernel(const float* __restrict__ audio, float2* __restrict__ mix,
                            float* __restrict__ mag, int T, int nT) {
  __shared__ float2 sh[N_FFT];
  __shared__ float2 tw[N_FFT / 2];
  int t = blockIdx.x, c = blockIdx.y, tid = threadIdx.x;
  init_twiddles(tw, tid, blockDim.x);
  const float* a = audio + (long)c * T;
  for (int n = tid; n < N_FFT; n += blockDim.x) {
    int j = brev12(n);
    long i = (long)t * HOP + j - (N_FFT / 2);
    if (i < 0) i = -i;                       // reflect pad (left)
    if (i >= T) i = 2L * T - 2 - i;          // reflect pad (right)
    sh[n] = make_float2(a[i] * hann_w(j), 0.0f);
  }
  __syncthreads();
  fft_lds(sh, tw, tid, blockDim.x, +1.0f);
  for (int f = tid; f < F_BINS; f += blockDim.x) {
    float2 v = sh[f];
    long o = ((long)c * F_BINS + f) * nT + t;
    mix[o] = v;
    mag[o] = sqrtf(v.x * v.x + v.y * v.y);
  }
}

// ------- 2) per-target linear + ReLU as bf16 WMMA GEMM --------------------
// out[t,g] = relu( sum_f mag[c][f][t] * W[s][g][f] ), stored V[t][g][c][s].
// Block tile 32(t) x 128(g), K-step 32, software-pipelined double-buffered
// LDS in WMMA fragment-ready layout:
//   frag index for lane (row r, half h): buf[(r*2 + h)*16 + e], e = lane's
//   element, with k = ((e>>3)<<4) + (e&7) + h*8  (CDNA5 16-bit A/B layout).
// Each lane then loads its 32B operand with two ds_load_b128.
__global__ void linear_wmma_kernel(const float* __restrict__ mag,
                                   const float* __restrict__ W,
                                   float* __restrict__ V, int nT) {
  __shared__ __align__(32) __bf16 Ab[2][32 * 32];    // [m-row][h][e]
  __shared__ __align__(32) __bf16 Bb[2][128 * 32];   // [n-row][h][e]
  int tid  = threadIdx.x;
  int wave = tid >> 5, lane = tid & 31;
  int half = lane >> 4, lm = lane & 15;
  int mi = wave >> 2, ni = wave & 3;          // 2x4 wave grid over 32x128 tile
  int n0 = blockIdx.x * 128, m0 = blockIdx.y * 32;
  int cs = blockIdx.z;
  int c = cs >> 2, s = cs & 3;
  const float* A  = mag + (long)c * F_BINS * nT;     // A[f][t], t contiguous
  const float* Bw = W + (long)s * F_BINS * F_BINS;   // B[g][f], f contiguous

  // staging maps (branchless clamp + select, coalesced global reads)
  int ka = tid >> 3, mb = (tid & 7) * 4;             // A: 4 consecutive t
  int ha = (ka >> 3) & 1, ea = (ka & 7) + ((ka >> 4) << 3);

  float aR[4];
  float bR[16];

  auto loadA = [&](int k0) {
    int kk = k0 + ka;
    const float* ap = A + (long)min(kk, F_BINS - 1) * nT;
    bool okk = kk < F_BINS;
#pragma unroll
    for (int j = 0; j < 4; ++j) {
      int mm = m0 + mb + j;
      float v = ap[min(mm, nT - 1)];
      aR[j] = (okk && mm < nT) ? v : 0.0f;
    }
  };
  auto storeA = [&](__bf16* buf) {
#pragma unroll
    for (int j = 0; j < 4; ++j)
      buf[((mb + j) * 2 + ha) * 16 + ea] = (__bf16)aR[j];
  };
  auto loadB = [&](int k0) {
#pragma unroll
    for (int i = 0; i < 2; ++i) {
      int q = tid + i * 256;                 // chunk: 8 consecutive f of row n
      int n = q >> 2, kb = q & 3;
      const float* bp = Bw + (long)min(n0 + n, F_BINS - 1) * F_BINS;
      bool okn = (n0 + n) < F_BINS;
#pragma unroll
      for (int j = 0; j < 8; ++j) {
        int kk = k0 + kb * 8 + j;
        float v = bp[min(kk, F_BINS - 1)];   // dword loads (row stride is odd)
        bR[i * 8 + j] = (okn && kk < F_BINS) ? v : 0.0f;
      }
    }
  };
  auto storeB = [&](__bf16* buf) {
#pragma unroll
    for (int i = 0; i < 2; ++i) {
      int q = tid + i * 256;
      int n = q >> 2, kb = q & 3;
      v8bf p;
#pragma unroll
      for (int j = 0; j < 8; ++j) p[j] = (__bf16)bR[i * 8 + j];
      *(v8bf*)&buf[(n * 2 + (kb & 1)) * 16 + (kb >> 1) * 8] = p;  // 16B store
    }
  };

  v8f acc0 = {}, acc1 = {};
  loadA(0);
  loadB(0);
  storeA(Ab[0]);
  storeB(Bb[0]);
  __syncthreads();

  for (int k0 = 0; k0 < F_BINS; k0 += 32) {
    int cur = (k0 >> 5) & 1, nxt = cur ^ 1;
    bool more = (k0 + 32) < F_BINS;
    if (more) { loadA(k0 + 32); loadB(k0 + 32); }   // global loads in flight
    // compute on current buffers: one A fragment feeds two B fragments
    v16bf av = *(const v16bf*)&Ab[cur][((mi * 16 + lm) * 2 + half) * 16];
    v16bf b0 = *(const v16bf*)&Bb[cur][((ni * 32 + lm) * 2 + half) * 16];
    v16bf b1 = *(const v16bf*)&Bb[cur][((ni * 32 + 16 + lm) * 2 + half) * 16];
    acc0 = __builtin_amdgcn_wmma_f32_16x16x32_bf16(false, av, false, b0,
                                                   (short)0, acc0, false, false);
    acc1 = __builtin_amdgcn_wmma_f32_16x16x32_bf16(false, av, false, b1,
                                                   (short)0, acc1, false, false);
    if (more) { storeA(Ab[nxt]); storeB(Bb[nxt]); } // fill other buffer
    __syncthreads();
  }

  // C/D layout: lane holds N = lm, rows M = r + half*8
  int g0 = n0 + ni * 32 + lm;
  int g1 = g0 + 16;
#pragma unroll
  for (int r = 0; r < 8; ++r) {
    int t = m0 + mi * 16 + half * 8 + r;
    if (t < nT) {
      long base = ((long)t * F_BINS) * 8 + c * 4 + s;
      if (g0 < F_BINS) {
        float v = acc0[r];
        V[base + (long)g0 * 8] = v > 0.0f ? v : 0.0f;
      }
      if (g1 < F_BINS) {
        float v = acc1[r];
        V[base + (long)g1 * 8] = v > 0.0f ? v : 0.0f;
      }
    }
  }
}

// -------------------- 3a) per-chunk max(|x|) ------------------------------
__global__ void chunk_max_kernel(const float2* __restrict__ mix,
                                 unsigned* __restrict__ mx, int nT) {
  __shared__ float red[256];
  int ck = blockIdx.x;
  long total = (long)WIN_LEN * F_BINS * N_CH;
  float m = 0.0f;
  for (long i = (long)blockIdx.y * blockDim.x + threadIdx.x; i < total;
       i += (long)gridDim.y * blockDim.x) {
    int c = (int)(i % N_CH);
    long rest = i / N_CH;
    int f = (int)(rest % F_BINS);
    int t = (int)(rest / F_BINS);
    float2 v = mix[((long)c * F_BINS + f) * nT + (ck * WIN_LEN + t)];
    m = fmaxf(m, sqrtf(v.x * v.x + v.y * v.y));
  }
  red[threadIdx.x] = m;
  __syncthreads();
  for (int o = 128; o > 0; o >>= 1) {
    if (threadIdx.x < o) red[threadIdx.x] = fmaxf(red[threadIdx.x], red[threadIdx.x + o]);
    __syncthreads();
  }
  if (threadIdx.x == 0) atomicMax(&mx[ck], __float_as_uint(red[0]));
}

// -------------------- 3b) Wiener EM (1 iteration) per (f, chunk) ----------
__global__ void wiener_kernel(const float2* __restrict__ mix, const float* __restrict__ V,
                              const unsigned* __restrict__ mx, float2* __restrict__ Y,
                              int nT) {
  const int NTH = 64;
  __shared__ float red[20][NTH];
  __shared__ float fin[20];
  int f = blockIdx.x, ck = blockIdx.y, tid = threadIdx.x;
  float max_abs = fmaxf(1.0f, __uint_as_float(mx[ck]) / SCALE_F);
  float inv_max = 1.0f / max_abs;

  // phase 1: accumulate R numerators (Hermitian 2x2 per source) and sum_t v
  float R00[4] = {}, R11[4] = {}, vs[4] = {};
  float2 R01[4] = {};
  for (int t = tid; t < WIN_LEN; t += NTH) {
    int tg = ck * WIN_LEN + t;
    float2 x0 = mix[(long)f * nT + tg];
    float2 x1 = mix[((long)F_BINS + f) * nT + tg];
    float a0 = sqrtf(x0.x * x0.x + x0.y * x0.y);
    float a1 = sqrtf(x1.x * x1.x + x1.y * x1.y);
    float2 p0 = a0 > 0.0f ? make_float2(x0.x / a0, x0.y / a0) : make_float2(1.0f, 0.0f);
    float2 p1 = a1 > 0.0f ? make_float2(x1.x / a1, x1.y / a1) : make_float2(1.0f, 0.0f);
    const float* vp = V + ((long)tg * F_BINS + f) * 8;
#pragma unroll
    for (int s = 0; s < 4; ++s) {
      float m0 = vp[s] * inv_max, m1 = vp[4 + s] * inv_max;  // |y0|,|y1|
      float2 y0 = make_float2(m0 * p0.x, m0 * p0.y);
      float2 y1 = make_float2(m1 * p1.x, m1 * p1.y);
      R00[s] += m0 * m0;
      R11[s] += m1 * m1;
      R01[s].x += y0.x * y1.x + y0.y * y1.y;   // y0*conj(y1)
      R01[s].y += y0.y * y1.x - y0.x * y1.y;
      vs[s] += 0.5f * (m0 * m0 + m1 * m1);
    }
  }
#pragma unroll
  for (int s = 0; s < 4; ++s) {
    red[s][tid] = R00[s];
    red[4 + s][tid] = R11[s];
    red[8 + 2 * s][tid] = R01[s].x;
    red[9 + 2 * s][tid] = R01[s].y;
    red[16 + s][tid] = vs[s];
  }
  __syncthreads();
  if (tid < 20) {
    float a = 0.0f;
    for (int i = 0; i < NTH; ++i) a += red[tid][i];
    fin[tid] = a;
  }
  __syncthreads();
  float nR00[4], nR11[4];
  float2 nR01[4];
#pragma unroll
  for (int s = 0; s < 4; ++s) {
    float d = 1.0f / (EPS_F + fin[16 + s]);
    nR00[s] = fin[s] * d;
    nR11[s] = fin[4 + s] * d;
    nR01[s] = make_float2(fin[8 + 2 * s] * d, fin[9 + 2 * s] * d);
  }

  // phase 2: per-frame gain and output
  for (int t = tid; t < WIN_LEN; t += NTH) {
    int tg = ck * WIN_LEN + t;
    float2 x0 = mix[(long)f * nT + tg];
    float2 x1 = mix[((long)F_BINS + f) * nT + tg];
    float2 xs0 = make_float2(x0.x * inv_max, x0.y * inv_max);
    float2 xs1 = make_float2(x1.x * inv_max, x1.y * inv_max);
    const float* vp = V + ((long)tg * F_BINS + f) * 8;
    float v[4];
#pragma unroll
    for (int s = 0; s < 4; ++s) {
      float m0 = vp[s] * inv_max, m1 = vp[4 + s] * inv_max;
      v[s] = 0.5f * (m0 * m0 + m1 * m1);
    }
    float C00 = SQRT_EPS, C11 = SQRT_EPS;
    float2 C01 = make_float2(0.0f, 0.0f);
#pragma unroll
    for (int s = 0; s < 4; ++s) {
      C00 += v[s] * nR00[s];
      C11 += v[s] * nR11[s];
      C01.x += v[s] * nR01[s].x;
      C01.y += v[s] * nR01[s].y;
    }
    float det = C00 * C11 - (C01.x * C01.x + C01.y * C01.y);  // Hermitian -> real det
    float id = 1.0f / det;
    float i00 = C11 * id, i11 = C00 * id;
    float2 i01 = make_float2(-C01.x * id, -C01.y * id);
    float2 i10 = make_float2(-C01.x * id, C01.y * id);
#pragma unroll
    for (int s = 0; s < 4; ++s) {
      float2 r01 = nR01[s];
      float2 r10 = make_float2(r01.x, -r01.y);
      float2 G00 = make_float2(nR00[s] * i00 + r01.x * i10.x - r01.y * i10.y,
                               r01.x * i10.y + r01.y * i10.x);
      float2 G01 = make_float2(nR00[s] * i01.x + r01.x * i11,
                               nR00[s] * i01.y + r01.y * i11);
      float2 G10 = make_float2(r10.x * i00 + nR11[s] * i10.x,
                               r10.y * i00 + nR11[s] * i10.y);
      float2 G11 = make_float2(r10.x * i01.x - r10.y * i01.y + nR11[s] * i11,
                               r10.x * i01.y + r10.y * i01.x);
      float sc = v[s] * max_abs;  // gain *= v; output *= max_abs
      float2 y0 = cmul(G00, xs0), t0 = cmul(G01, xs1);
      float2 y1 = cmul(G10, xs0), t1 = cmul(G11, xs1);
      y0.x = (y0.x + t0.x) * sc; y0.y = (y0.y + t0.y) * sc;
      y1.x = (y1.x + t1.x) * sc; y1.y = (y1.y + t1.y) * sc;
      Y[((long)(s * 2 + 0) * nT + tg) * F_BINS + f] = y0;
      Y[((long)(s * 2 + 1) * nT + tg) * F_BINS + f] = y1;
    }
  }
}

// -------------------- 4) iFFT + windowed overlap-add (atomic) -------------
__global__ void istft_kernel(const float2* __restrict__ Y, float* __restrict__ acc,
                             int nT, int totalLen) {
  __shared__ float2 sh[N_FFT];
  __shared__ float2 tw[N_FFT / 2];
  int t = blockIdx.x, sc = blockIdx.y, tid = threadIdx.x;
  init_twiddles(tw, tid, blockDim.x);
  const float2* Yp = Y + ((long)sc * nT + t) * F_BINS;
  for (int n = tid; n < N_FFT; n += blockDim.x) {
    int j = brev12(n);
    float2 v;
    if (j < F_BINS) v = Yp[j];
    else {
      float2 u = Yp[N_FFT - j];
      v = make_float2(u.x, -u.y);   // conjugate symmetry
    }
    sh[n] = v;
  }
  __syncthreads();
  fft_lds(sh, tw, tid, blockDim.x, -1.0f);   // inverse (unscaled)
  float* ap = acc + (long)sc * totalLen + (long)t * HOP;
  const float inv_n = 1.0f / (float)N_FFT;
  for (int n = tid; n < N_FFT; n += blockDim.x) {
    atomicAdd(&ap[n], sh[n].x * inv_n * hann_w(n));
  }
}

// -------------------- 5) normalize by analytic wsum, crop pad -------------
__global__ void norm_kernel(const float* __restrict__ acc, float* __restrict__ out,
                            int T, int nT, int totalLen) {
  long i = (long)blockIdx.x * blockDim.x + threadIdx.x;
  long tot = (long)N_SRC * N_CH * T;
  if (i >= tot) return;
  int tau = (int)(i % T);
  int sc = (int)(i / T);
  int p = tau + N_FFT / 2;
  int tmin = max(0, (p - (N_FFT - 1) + HOP - 1) / HOP);
  int tmax = min(nT - 1, p / HOP);
  float wsum = 0.0f;
  for (int tt = tmin; tt <= tmax; ++tt) {
    float w = hann_w(p - tt * HOP);
    wsum += w * w;
  }
  float denom = wsum > 1e-11f ? wsum : 1.0f;
  out[i] = acc[(long)sc * totalLen + p] / denom;
}

extern "C" void kernel_launch(void* const* d_in, const int* in_sizes, int n_in,
                              void* d_out, int out_size, void* d_ws, size_t ws_size,
                              hipStream_t stream) {
  (void)n_in; (void)out_size; (void)ws_size;
  const float* audio = (const float*)d_in[0];
  const float* W     = (const float*)d_in[1];
  int T  = in_sizes[0] / N_CH;          // B=1
  int nT = T / HOP + 1;                 // 1500
  int nck = nT / WIN_LEN;               // 5
  int totalLen = HOP * (nT - 1) + N_FFT;

  char* ws = (char*)d_ws;
  size_t off = 0;
  float2*  mix  = (float2*)(ws + off); off += (size_t)N_CH * F_BINS * nT * sizeof(float2);
  float*   mag  = (float*)(ws + off);  off += (size_t)N_CH * F_BINS * nT * sizeof(float);
  float*   V    = (float*)(ws + off);  off += (size_t)nT * F_BINS * N_CH * N_SRC * sizeof(float);
  float2*  Y    = (float2*)(ws + off); off += (size_t)N_SRC * N_CH * nT * F_BINS * sizeof(float2);
  float*   acc  = (float*)(ws + off);  off += (size_t)N_SRC * N_CH * totalLen * sizeof(float);
  unsigned* mx  = (unsigned*)(ws + off); off += 64;

  hipMemsetAsync(acc, 0, (size_t)N_SRC * N_CH * totalLen * sizeof(float), stream);
  hipMemsetAsync(mx, 0, 64, stream);
  if (nck * WIN_LEN < nT)  // frames not covered by Wiener chunking must be zero
    hipMemsetAsync(Y, 0, (size_t)N_SRC * N_CH * nT * F_BINS * sizeof(float2), stream);

  stft_kernel<<<dim3(nT, N_CH), 256, 0, stream>>>(audio, mix, mag, T, nT);
  linear_wmma_kernel<<<dim3((F_BINS + 127) / 128, (nT + 31) / 32, N_CH * N_SRC),
                       256, 0, stream>>>(mag, W, V, nT);
  chunk_max_kernel<<<dim3(nck, 16), 256, 0, stream>>>(mix, mx, nT);
  wiener_kernel<<<dim3(F_BINS, nck), 64, 0, stream>>>(mix, V, mx, Y, nT);
  istft_kernel<<<dim3(nT, N_SRC * N_CH), 256, 0, stream>>>(Y, acc, nT, totalLen);
  long tot = (long)N_SRC * N_CH * T;
  norm_kernel<<<dim3((unsigned)((tot + 255) / 256)), 256, 0, stream>>>(
      acc, (float*)d_out, T, nT, totalLen);
}
